// MaskOutput_44916767981924
// MI455X (gfx1250) — compile-verified, matrix operand order checked by reading, not run on hardware
//
#include <hip/hip_runtime.h>
#include <hip/hip_bf16.h>

typedef __attribute__((ext_vector_type(16))) _Float16 v16h;
typedef __attribute__((ext_vector_type(8)))  float    v8f;

#define INF_VAL 100000000.0f
#define HH 800
#define WW 1024
#define BB 16
#define GG 48
#define NUMC 80
#define NLOC 17064           // 12800+3200+800+208+56
#define TOT  (BB * NLOC)     // 273024 output rows

// ---------------- pass 0: zero cnt scratch (B*G floats) ----------------
__global__ void zero_cnt_kernel(float* __restrict__ cnt) {
    int i = blockIdx.x * blockDim.x + threadIdx.x;
    if (i < BB * GG) cnt[i] = 0.0f;
}

// ---------------- pass 1: cnt[b][g] = sum_loc is_in  via WMMA ----------------
// A (16x32 f16): M = gt within tile, K = location within 32-chunk.
// 16-bit A layout: lane L holds row M=L%16; lanes 0-15: K in {0..7, 16..23};
// lanes 16-31: K in {8..15, 24..31}; VGPR j packs 2 consecutive K.
// B = all-ones (32x16) -> D[m][n] = sum_k A[m][k]; every column equals cnt_m.
#define CHUNKS_TOTAL 534     // ceil(17064/32); c1-c0 always even (64 or 22)
#define CHUNKS_PER   64
#define GROUPS       9       // ceil(534/64)

__device__ __forceinline__ v16h build_a(float xl, float yl, int base, int khi,
                                        float bx1, float by1, float bx2, float by2) {
    v16h a;
    #pragma unroll
    for (int e = 0; e < 16; ++e) {
        const int j = e >> 1, p = e & 1;
        const int k = ((j < 4) ? (2 * j + p) : (16 + 2 * (j - 4) + p)) + khi;
        // pull location (x,y) for lane k of this chunk via lane permute
        const float xs = __shfl(xl, k);
        const float ys = __shfl(yl, k);
        const float le = xs - bx1, tp = ys - by1, rr = bx2 - xs, bo = by2 - ys;
        const float mn = fminf(fminf(le, tp), fminf(rr, bo));
        const float v  = (mn > 0.0f && (base + k) < NLOC) ? 1.0f : 0.0f;
        a[e] = (_Float16)v;
    }
    return a;
}

__global__ __launch_bounds__(32)
void count_kernel(const float* __restrict__ locations,
                  const float* __restrict__ gt_boxes,
                  float* __restrict__ cnt) {
    const int bid  = blockIdx.x;              // B * 3 * GROUPS blocks, 1 wave each
    const int b    = bid / (3 * GROUPS);
    const int rem  = bid % (3 * GROUPS);
    const int tile = rem / GROUPS;            // which 16-GT tile (0..2)
    const int grp  = rem % GROUPS;            // which chunk-group
    const int lane = threadIdx.x;
    const int gt   = tile * 16 + (lane & 15); // A-matrix row M handled by this lane

    const float bx1 = gt_boxes[(b * GG + gt) * 4 + 0];
    const float by1 = gt_boxes[(b * GG + gt) * 4 + 1];
    const float bx2 = gt_boxes[(b * GG + gt) * 4 + 2];
    const float by2 = gt_boxes[(b * GG + gt) * 4 + 3];

    v16h ones;
    #pragma unroll
    for (int e = 0; e < 16; ++e) ones[e] = (_Float16)1.0f;

    v8f c = {};
    const int khi = (lane < 16) ? 0 : 8;      // lane-group K offset
    const int c0 = grp * CHUNKS_PER;
    const int c1 = (c0 + CHUNKS_PER < CHUNKS_TOTAL) ? (c0 + CHUNKS_PER) : CHUNKS_TOTAL;

    // uniform bounds -> EXEC all-1s at every WMMA; c1-c0 is even, unroll x2
    for (int ch = c0; ch < c1; ch += 2) {
        const int base0 = ch * 32;
        const int base1 = base0 + 32;
        // one coalesced b64 load per lane per chunk (clamped, branchless tail)
        int l0 = base0 + lane; l0 = l0 < NLOC ? l0 : NLOC - 1;
        int l1 = base1 + lane; l1 = l1 < NLOC ? l1 : NLOC - 1;
        const float xl0 = locations[2 * l0 + 0];
        const float yl0 = locations[2 * l0 + 1];
        const float xl1 = locations[2 * l1 + 0];
        const float yl1 = locations[2 * l1 + 1];

        const v16h a0 = build_a(xl0, yl0, base0, khi, bx1, by1, bx2, by2);
        c = __builtin_amdgcn_wmma_f32_16x16x32_f16(
                false, a0, false, ones, (short)0, c, false, false);
        const v16h a1 = build_a(xl1, yl1, base1, khi, bx1, by1, bx2, by2);
        c = __builtin_amdgcn_wmma_f32_16x16x32_f16(
                false, a1, false, ones, (short)0, c, false, false);
    }

    // D layout: VGPR r -> (lanes 0-15: M=r, N=lane), (lanes 16-31: M=8+r, N=lane-16)
    if (lane == 0) {
        #pragma unroll
        for (int r = 0; r < 8; ++r)
            atomicAdd(&cnt[b * GG + tile * 16 + r], c[r]);
    } else if (lane == 16) {
        #pragma unroll
        for (int r = 0; r < 8; ++r)
            atomicAdd(&cnt[b * GG + tile * 16 + 8 + r], c[r]);
    }
}

// ---------------- pass 2: per-row assignment ----------------
__global__ __launch_bounds__(256)
void assign_kernel(const float* __restrict__ locations,
                   const float* __restrict__ gt_boxes,
                   const int*   __restrict__ gt_classes,
                   const float* __restrict__ gt_masks,
                   const float* __restrict__ cnt,
                   float* __restrict__ out) {
    const int t = blockIdx.x * 256 + threadIdx.x;   // output row (level-major)
    if (t >= TOT) return;

    const int   nl[5]   = {12800, 3200, 800, 208, 56};
    const int   beg[5]  = {0, 12800, 16000, 16800, 17008};
    const float invs[5] = {1.0f/8, 1.0f/16, 1.0f/32, 1.0f/64, 1.0f/128};
    const float lo[5]   = {-1.0f, 64.0f, 128.0f, 256.0f, 512.0f};
    const float hi[5]   = {64.0f, 128.0f, 256.0f, 512.0f, INF_VAL};

    int lv = 0;
    #pragma unroll
    for (int i = 1; i < 5; ++i) if (t >= BB * beg[i]) lv = i;
    const int u = t - BB * beg[lv];
    const int b = u / nl[lv];
    const int j = u - b * nl[lv];
    const int n = beg[lv] + j;                       // global location index

    const float x = locations[2 * n + 0];
    const float y = locations[2 * n + 1];
    int xi = (int)x; xi = xi < 0 ? 0 : (xi > WW - 1 ? WW - 1 : xi);
    int yi = (int)y; yi = yi < 0 ? 0 : (yi > HH - 1 ? HH - 1 : yi);
    const float mask_loc = gt_masks[(b * HH + yi) * WW + xi];
    const bool  mpos = mask_loc > 0.0f;

    float best_area = INF_VAL;
    int   best_g = 0;
    unsigned long long inbits = 0ull;

    for (int g = 0; g < GG; ++g) {
        const float* bx = &gt_boxes[(b * GG + g) * 4];
        const float x1 = bx[0], y1 = bx[1], x2 = bx[2], y2 = bx[3];
        const float le = x - x1, tp = y - y1, rr = x2 - x, bo = y2 - y;
        const float mn = fminf(fminf(le, tp), fminf(rr, bo));
        const float mx = fmaxf(fmaxf(le, tp), fmaxf(rr, bo));
        const bool in_raw = mn > 0.0f;
        const bool gated  = (cnt[b * GG + g] > 1.0f) ? (in_raw && mpos) : in_raw;
        if (gated) inbits |= (1ull << g);
        const bool cared = (mx >= lo[lv]) && (mx <= hi[lv]);
        const float area = (gated && cared) ? (x2 - x1) * (y2 - y1) : INF_VAL;
        if (area < best_area) { best_area = area; best_g = g; }  // first-min tiebreak
    }

    int label = gt_classes[b * GG + best_g];
    if (best_area >= INF_VAL) label = NUMC;
    if (mask_loc == 0.0f) label = NUMC;

    const float* bb = &gt_boxes[(b * GG + best_g) * 4];
    const float s  = invs[lv];
    const float rl = (x - bb[0]) * s, rt = (y - bb[1]) * s;
    const float rrr = (bb[2] - x) * s, rb = (bb[3] - y) * s;
    const int tind = (best_g + b * GG) * (int)mask_loc;
    const float sc = (label == 0) ? 1.0f : 0.0f;

    out[t] = (float)label;                            // labels
    float* rg = out + TOT + 4 * t;                    // reg
    rg[0] = rl; rg[1] = rt; rg[2] = rrr; rg[3] = rb;
    out[5 * TOT + t] = (float)tind;                   // tinds
    float* ii = out + 6 * TOT + (long long)t * GG;    // is_in
    #pragma unroll 8
    for (int g = 0; g < GG; ++g)
        ii[g] = ((inbits >> g) & 1ull) ? sc : 0.0f;
}

extern "C" void kernel_launch(void* const* d_in, const int* in_sizes, int n_in,
                              void* d_out, int out_size, void* d_ws, size_t ws_size,
                              hipStream_t stream) {
    const float* locations  = (const float*)d_in[0];
    const float* gt_boxes   = (const float*)d_in[1];
    const int*   gt_classes = (const int*)  d_in[2];
    const float* gt_masks   = (const float*)d_in[3];
    float*       out        = (float*)d_out;
    float*       cnt        = (float*)d_ws;          // B*G = 768 floats

    zero_cnt_kernel<<<(BB * GG + 255) / 256, 256, 0, stream>>>(cnt);
    count_kernel<<<BB * 3 * GROUPS, 32, 0, stream>>>(locations, gt_boxes, cnt);
    assign_kernel<<<(TOT + 255) / 256, 256, 0, stream>>>(
        locations, gt_boxes, gt_classes, gt_masks, cnt, out);
}
